// FFSNN_v2_87050397155437
// MI455X (gfx1250) — matrix operand order repaired and checked by reading, not compiled, not run on hardware
//
#include <hip/hip_runtime.h>

// ---------------------------------------------------------------------------
// FFSNN fused forward for MI455X (gfx1250, wave32, WMMA)
//   B=1024, T=196, H=512, OUT=10
//   64 blocks x 512 threads (16 waves); each block owns 16 batch rows for all
//   196 steps; each wave owns 32 neurons per layer (2 WMMA n-tiles).
//   Sized so per-wave live state (~130 VGPR) never spills (round-1 lesson).
// ---------------------------------------------------------------------------

#define T_STEPS 196
#define BT      16          // batch rows per block
#define H       512
#define RS      520         // LDS spike-row stride in halves (pad vs bank conflicts)
#define THRESH  0.5f
#define DECAY   0.2f

typedef _Float16 v16h __attribute__((ext_vector_type(16)));
typedef _Float16 v8h  __attribute__((ext_vector_type(8)));
typedef float    v8f  __attribute__((ext_vector_type(8)));
typedef float    v4f  __attribute__((ext_vector_type(4)));
typedef unsigned int u32x4 __attribute__((ext_vector_type(4)));
typedef int          i32x4 __attribute__((ext_vector_type(4)));
typedef int          i32x8 __attribute__((ext_vector_type(8)));

__device__ __forceinline__ v8f wmma16(v16h a, v16h b, v8f c) {
  // D = A(16x32 f16) x B(32x16 f16) + C(16x16 f32)
  return __builtin_amdgcn_wmma_f32_16x16x32_f16(false, a, false, b, (short)0, c,
                                                false, false);
}

// A-fragment (16x32 f16) from row-major LDS spike buffer.
// ISA layout: lane L holds row M=L&15; half i -> K = (i&7) + 16*(i>>3) + 8*(L>>4).
// => two 16B ds_load_b128 per lane at K0 and K0+16.
__device__ __forceinline__ v16h load_a_frag(const _Float16* sIn, int lane, int kTile) {
  const int m  = lane & 15;
  const int k0 = kTile + ((lane >> 4) << 3);
  const _Float16* p = sIn + m * RS + k0;
  v8h lo = *(const v8h*)(p);
  v8h hi = *(const v8h*)(p + 16);
  v16h r;
#pragma unroll
  for (int i = 0; i < 8; ++i) { r[i] = lo[i]; r[i + 8] = hi[i]; }
  return r;
}

// One hidden layer for this wave's 2 n-tiles: z = spikes(sIn) @ W^T + b via WMMA,
// membrane update (recompute previous spike from membrane), emit new spikes to sOut.
__device__ __forceinline__ void layer_wmma(const _Float16* sIn, _Float16* sOut,
                                           const v16h* __restrict__ frag,  // wave-offset base
                                           const float* bL, v8f* m,
                                           int lane, int ntBase) {
  v8f acc[2];
#pragma unroll
  for (int j = 0; j < 2; ++j) {
#pragma unroll
    for (int e = 0; e < 8; ++e) acc[j][e] = bL[j];   // bias depends only on N
  }
  // Bounded unroll: keeps in-flight B-fragments small -> no spills.
#pragma unroll 4
  for (int ks = 0; ks < 16; ++ks) {
    v16h a  = load_a_frag(sIn, lane, ks * 32);
    v16h b0 = frag[(0 * 16 + ks) * 32 + lane];       // 32B/lane, 1KB contiguous frag
    v16h b1 = frag[(1 * 16 + ks) * 32 + lane];
    acc[0] = wmma16(a, b0, acc[0]);                  // 2 independent WMMA chains
    acc[1] = wmma16(a, b1, acc[1]);
  }
  const int rowOff = (lane >> 4) << 3;
#pragma unroll
  for (int j = 0; j < 2; ++j) {
    v8f mo = m[j];
#pragma unroll
    for (int r = 0; r < 8; ++r) {
      float dec = (mo[r] > THRESH) ? 0.0f : mo[r] * DECAY;  // m*decay*(1-spike)
      mo[r] = dec + acc[j][r];
    }
    m[j] = mo;
    const int col = (ntBase + j) * 16 + (lane & 15);
#pragma unroll
    for (int r = 0; r < 8; ++r)
      sOut[(r + rowOff) * RS + col] = (mo[r] > THRESH) ? (_Float16)1.0f
                                                       : (_Float16)0.0f;
  }
}

// ---------------------------------------------------------------------------
// Prep: pack W2/W3/W4 (fp32 row-major [n][k]) into f16 B-fragment order:
//   half index = nt*8192 + ks*512 + lane*16 + i
//   value      = W[n = nt*16 + (lane&15)][k = ks*32 + 8*(lane>>4) + (i&7) + 16*(i>>3)]
// W4 padded from 10 to 16 output rows with zeros.
// ---------------------------------------------------------------------------
__global__ void ffsnn_pack_kernel(const float* __restrict__ W2,
                                  const float* __restrict__ W3,
                                  const float* __restrict__ W4,
                                  _Float16* __restrict__ f2,
                                  _Float16* __restrict__ f3,
                                  _Float16* __restrict__ f4) {
  const int tid  = blockIdx.x * 256 + threadIdx.x;
  const int i    = tid & 15;
  const int lane = (tid >> 4) & 31;
  const int ks   = (tid >> 9) & 15;
  const int k    = ks * 32 + ((lane >> 4) << 3) + (i & 7) + ((i >> 3) << 4);
  if (tid < 262144) {
    const int nt = tid >> 13;
    const int n  = nt * 16 + (lane & 15);
    f2[tid] = (_Float16)W2[n * H + k];
  } else if (tid < 524288) {
    const int t2 = tid - 262144;
    const int nt = t2 >> 13;
    const int n  = nt * 16 + (lane & 15);
    f3[t2] = (_Float16)W3[n * H + k];
  } else if (tid < 532480) {
    const int t4 = tid - 524288;             // 16 ks * 32 lanes * 16 halves
    const int n  = lane & 15;
    f4[t4] = (n < 10) ? (_Float16)W4[n * H + k] : (_Float16)0.0f;
  }
}

// ---------------------------------------------------------------------------
// Main fused SNN kernel: 512 threads = 16 waves; wave w owns n-tiles {2w,2w+1}.
// ---------------------------------------------------------------------------
__global__ __launch_bounds__(512, 1)
void ffsnn_main_kernel(const float* __restrict__ x,
                       const float* __restrict__ W1, const float* __restrict__ b1,
                       const float* __restrict__ b2, const float* __restrict__ b3,
                       const float* __restrict__ b4,
                       const v16h* __restrict__ f2, const v16h* __restrict__ f3,
                       const v16h* __restrict__ f4,
                       float* __restrict__ out) {
  extern __shared__ char smem[];
  float*    sX  = (float*)smem;                       // 16*784 f32 = 50176 B (offset 0)
  _Float16* sA1 = (_Float16*)(smem + 50176);          // 16*RS halves each
  _Float16* sA2 = sA1 + 16 * RS;
  _Float16* sA3 = sA2 + 16 * RS;

  const int lane    = threadIdx.x & 31;
  const int w       = threadIdx.x >> 5;               // 0..15
  const int ntBase  = w * 2;                          // this wave's first n-tile
  const int rowBase = blockIdx.x * BT;
  const int rowOff  = (lane >> 4) << 3;               // 0 or 8 (C/D layout M offset)

  // ---- Load this block's 16x784 x-slice into LDS via the Tensor Data Mover ----
#if __has_builtin(__builtin_amdgcn_tensor_load_to_lds)
  if (w == 0) {
    unsigned long long ga =
        (unsigned long long)(size_t)(x + (size_t)rowBase * 784);
    u32x4 g0; i32x8 g1; i32x4 g2 = {0, 0, 0, 0}, g3 = {0, 0, 0, 0};
    g0[0] = 1u;                                        // count=1 (valid D#)
    g0[1] = 0u;                                        // lds_addr = 0 (sX at LDS base)
    g0[2] = (unsigned)(ga & 0xFFFFFFFFu);              // global_addr[31:0]
    g0[3] = (unsigned)((ga >> 32) & 0x01FFFFFFu) | 0x80000000u; // addr[56:32] | type=2
    g1[0] = (int)(2u << 16);                           // data_size = 4 bytes
    g1[1] = (int)(784u << 16);                         // tensor_dim0[15:0] in [63:48]
    g1[2] = (int)(1024u << 16);                        // dim0 hi=0 | tensor_dim1[15:0]
    g1[3] = (int)(784u << 16);                         // dim1 hi=0 | tile_dim0=784
    g1[4] = 16;                                        // tile_dim1=16, tile_dim2=0
    g1[5] = 784;                                       // tensor_dim0_stride lo
    g1[6] = 0;
    g1[7] = 0;
#if defined(__clang_major__) && (__clang_major__ >= 23)
    i32x8 gx = {0, 0, 0, 0, 0, 0, 0, 0};
    __builtin_amdgcn_tensor_load_to_lds(g0, g1, g2, g3, gx, 0);
#else
    __builtin_amdgcn_tensor_load_to_lds(g0, g1, g2, g3, 0);
#endif
    __builtin_amdgcn_s_wait_tensorcnt(0);
  }
#else
  for (int idx = threadIdx.x; idx < BT * 784; idx += 512)
    sX[idx] = x[(size_t)(rowBase + idx / 784) * 784 + (idx % 784)];
#endif
  __syncthreads();

  // ---- Per-wave resident constants: W1 slice + biases for owned neurons ----
  v4f   w1L[2];
  float b1L[2], b2L[2], b3L[2];
#pragma unroll
  for (int j = 0; j < 2; ++j) {
    const int n = (ntBase + j) * 16 + (lane & 15);
    w1L[j] = *(const v4f*)(W1 + n * 4);
    b1L[j] = b1[n];
    b2L[j] = b2[n];
    b3L[j] = b3[n];
  }

  v8f m1[2], m2[2], m3[2], acc4;
#pragma unroll
  for (int j = 0; j < 2; ++j)
#pragma unroll
    for (int e = 0; e < 8; ++e) { m1[j][e] = 0.f; m2[j][e] = 0.f; m3[j][e] = 0.f; }
#pragma unroll
  for (int e = 0; e < 8; ++e) acc4[e] = 0.f;

  const v16h* f2w = f2 + ntBase * 16 * 32;
  const v16h* f3w = f3 + ntBase * 16 * 32;

  for (int t = 0; t < T_STEPS; ++t) {
    // Reference indexing quirk: only t<48 slices fresh columns; rest reuse 780..783.
    const int ct = (t < 48) ? (t * 4) : 780;

    // ---- Layer 1: K=4 GEMM on VALU (x from LDS, broadcast reads) ----
#pragma unroll
    for (int j = 0; j < 2; ++j) {
      v8f mo = m1[j];
      const int col = (ntBase + j) * 16 + (lane & 15);
#pragma unroll
      for (int r = 0; r < 8; ++r) {
        v4f xr = *(const v4f*)(sX + (r + rowOff) * 784 + ct);
        float z = b1L[j] + xr[0] * w1L[j][0] + xr[1] * w1L[j][1]
                         + xr[2] * w1L[j][2] + xr[3] * w1L[j][3];
        float dec = (mo[r] > THRESH) ? 0.0f : mo[r] * DECAY;
        mo[r] = dec + z;
      }
      m1[j] = mo;
#pragma unroll
      for (int r = 0; r < 8; ++r)
        sA1[(r + rowOff) * RS + col] = (mo[r] > THRESH) ? (_Float16)1.0f
                                                        : (_Float16)0.0f;
    }
    __syncthreads();

    // ---- Layer 2 / Layer 3: WMMA ----
    layer_wmma(sA1, sA2, f2w, b2L, m2, lane, ntBase);
    __syncthreads();
    layer_wmma(sA2, sA3, f3w, b3L, m3, lane, ntBase);
    __syncthreads();

    // ---- Layer 4: one padded 16x16 output tile, wave 0, accumulates over t ----
    if (w == 0) {
#pragma unroll 4
      for (int ks = 0; ks < 16; ++ks) {
        v16h a = load_a_frag(sA3, lane, ks * 32);
        acc4 = wmma16(a, f4[ks * 32 + lane], acc4);
      }
    }
  }

  // ---- Epilogue: out = acc/T + b4 ----
  if (w == 0) {
    const int o = lane & 15;
    if (o < 10) {
      const float bb = b4[o];
#pragma unroll
      for (int r = 0; r < 8; ++r)
        out[(size_t)(rowBase + r + rowOff) * 10 + o] =
            acc4[r] * (1.0f / (float)T_STEPS) + bb;
    }
  }
}

// ---------------------------------------------------------------------------
extern "C" void kernel_launch(void* const* d_in, const int* in_sizes, int n_in,
                              void* d_out, int out_size, void* d_ws, size_t ws_size,
                              hipStream_t stream) {
  const float* x  = (const float*)d_in[0];
  const float* W1 = (const float*)d_in[1];
  const float* b1 = (const float*)d_in[2];
  const float* W2 = (const float*)d_in[3];
  const float* b2 = (const float*)d_in[4];
  const float* W3 = (const float*)d_in[5];
  const float* b3 = (const float*)d_in[6];
  const float* W4 = (const float*)d_in[7];
  const float* b4 = (const float*)d_in[8];

  _Float16* f2 = (_Float16*)d_ws;          // 512*512 halves = 512 KB
  _Float16* f3 = f2 + 262144;              // 512 KB
  _Float16* f4 = f3 + 262144;              // 16 KB (padded 16x512)

  ffsnn_pack_kernel<<<(532480 + 255) / 256, 256, 0, stream>>>(W2, W3, W4, f2, f3, f4);

  const size_t lds = 50176 + 3 * (size_t)(16 * RS) * sizeof(_Float16); // ~100 KB
  ffsnn_main_kernel<<<64, 512, lds, stream>>>(
      x, W1, b1, b2, b3, b4,
      (const v16h*)f2, (const v16h*)f3, (const v16h*)f4,
      (float*)d_out);
}
